// FP8DynamicLinear_68083821576376
// MI455X (gfx1250) — 8-bit matrix operand order repaired and checked by reading, compiled, not run on hardware
//
#include <hip/hip_runtime.h>
#include <stdint.h>

typedef int   v4i  __attribute__((ext_vector_type(4)));
typedef int   v16i __attribute__((ext_vector_type(16)));
typedef float v8f  __attribute__((ext_vector_type(8)));

#define FP8_MAX_F 448.0f

#if __has_builtin(__builtin_amdgcn_global_load_async_to_lds_b128) && \
    __has_builtin(__builtin_amdgcn_s_wait_asynccnt)
#define USE_ASYNC_COPY 1
typedef __attribute__((address_space(1))) v4i glb_v4i;
typedef __attribute__((address_space(3))) v4i lds_v4i;
#else
#define USE_ASYNC_COPY 0
#endif

__device__ __forceinline__ void copy16_to_lds(const unsigned char* g,
                                              unsigned char* l) {
#if USE_ASYNC_COPY
    __builtin_amdgcn_global_load_async_to_lds_b128((glb_v4i*)g, (lds_v4i*)l,
                                                   0, 0);
#else
    *(uint4*)l = *(const uint4*)g;
#endif
}

__device__ __forceinline__ void copy_fence() {
#if USE_ASYNC_COPY
    __builtin_amdgcn_s_wait_asynccnt(0);
#endif
}

__global__ void amax_kernel(const float* __restrict__ x, int n4,
                            int* __restrict__ amax_bits) {
    __shared__ float red[256];
    const int tid    = threadIdx.x;
    const int idx    = blockIdx.x * blockDim.x + tid;
    const int stride = gridDim.x * blockDim.x;
    const float4* x4 = (const float4*)x;
    float m = 0.0f;
    for (int i = idx; i < n4; i += stride) {
        float4 v = x4[i];
        m = fmaxf(m, fmaxf(fmaxf(fabsf(v.x), fabsf(v.y)),
                           fmaxf(fabsf(v.z), fabsf(v.w))));
    }
    red[tid] = m;
    __syncthreads();
    for (int s = 128; s > 0; s >>= 1) {
        if (tid < s) red[tid] = fmaxf(red[tid], red[tid + s]);
        __syncthreads();
    }
    if (tid == 0) atomicMax(amax_bits, __float_as_int(red[0]));
}

__global__ void scale_kernel(const int* __restrict__ amax_bits,
                             float* __restrict__ scales) {
    float amax = fmaxf(__int_as_float(*amax_bits), 1e-12f);
    scales[0] = FP8_MAX_F / amax;
    scales[1] = amax / FP8_MAX_F;
}

__device__ __forceinline__ unsigned int f32_to_e4m3_sw(float f) {
    unsigned int sign = (__float_as_uint(f) >> 24) & 0x80u;
    float a = fabsf(f);
    unsigned int code;
    if (a < 0.015625f) {
        code = (unsigned int)rintf(a * 512.0f);
    } else {
        unsigned int b = __float_as_uint(a);
        unsigned int r = b + 0x0007FFFFu + ((b >> 20) & 1u);
        int e = (int)(r >> 23) - 127;
        unsigned int mant = (r >> 20) & 7u;
        code = (unsigned int)((e + 7) << 3) | mant;
        if (code > 0x7Eu) code = 0x7Eu;
    }
    return sign | code;
}

__device__ __forceinline__ unsigned int pack4_e4m3(float a, float b,
                                                   float c, float d) {
#if __has_builtin(__builtin_amdgcn_cvt_pk_fp8_f32)
    int r = __builtin_amdgcn_cvt_pk_fp8_f32(a, b, 0, false);
    r     = __builtin_amdgcn_cvt_pk_fp8_f32(c, d, r, true);
    return (unsigned int)r;
#else
    return f32_to_e4m3_sw(a) | (f32_to_e4m3_sw(b) << 8) |
           (f32_to_e4m3_sw(c) << 16) | (f32_to_e4m3_sw(d) << 24);
#endif
}

__global__ void quant_kernel(const float* __restrict__ src,
                             unsigned int* __restrict__ dst, int n4,
                             const float* __restrict__ scales, int use_scale) {
    const int idx    = blockIdx.x * blockDim.x + threadIdx.x;
    const int stride = gridDim.x * blockDim.x;
    const float qs   = use_scale ? scales[0] : 1.0f;
    const float4* s4 = (const float4*)src;
    for (int i = idx; i < n4; i += stride) {
        float4 v = s4[i];
        float a = fminf(fmaxf(v.x * qs, -FP8_MAX_F), FP8_MAX_F);
        float b = fminf(fmaxf(v.y * qs, -FP8_MAX_F), FP8_MAX_F);
        float c = fminf(fmaxf(v.z * qs, -FP8_MAX_F), FP8_MAX_F);
        float d = fminf(fmaxf(v.w * qs, -FP8_MAX_F), FP8_MAX_F);
        dst[i] = pack4_e4m3(a, b, c, d);
    }
}

#define BLK_K      128
#define LDS_STRIDE 144
#define TILE_BYTES (128 * LDS_STRIDE)

__global__ __launch_bounds__(256)
void fp8_gemm_wmma_kernel(const unsigned char* __restrict__ qx,
                          const unsigned char* __restrict__ qw,
                          const float* __restrict__ scales,
                          const float* __restrict__ wscale,
                          const float* __restrict__ bias,
                          float* __restrict__ out,
                          int M, int N, int K) {
    __shared__ __align__(16) unsigned char smem[4 * TILE_BYTES];

    const int tid  = threadIdx.x;
    const int lane = tid & 31;
    const int wave = tid >> 5;
    const int half = lane >> 4;
    const int lrow = lane & 15;
    const int wm   = wave >> 1;
    const int wn   = wave & 1;

    const int mblk = blockIdx.y * 128;
    const int nblk = blockIdx.x * 128;

    const int crow0 = tid >> 3;
    const int cseg  = tid & 7;

    const unsigned char* gA = qx + (size_t)mblk * K;
    const unsigned char* gB = qw + (size_t)nblk * K;

    v8f acc[8] = {};

    int cur = 0;
#pragma unroll
    for (int r = 0; r < 4; ++r) {
        const int row = crow0 + 32 * r;
        copy16_to_lds(gA + (size_t)row * K + cseg * 16,
                      &smem[0 * TILE_BYTES + row * LDS_STRIDE + cseg * 16]);
        copy16_to_lds(gB + (size_t)row * K + cseg * 16,
                      &smem[2 * TILE_BYTES + row * LDS_STRIDE + cseg * 16]);
    }

    for (int kc = 0; kc < K; kc += BLK_K) {
        copy_fence();
        __syncthreads();

        if (kc + BLK_K < K) {
            const int nb = cur ^ 1;
#pragma unroll
            for (int r = 0; r < 4; ++r) {
                const int row = crow0 + 32 * r;
                copy16_to_lds(
                    gA + (size_t)row * K + kc + BLK_K + cseg * 16,
                    &smem[nb * TILE_BYTES + row * LDS_STRIDE + cseg * 16]);
                copy16_to_lds(
                    gB + (size_t)row * K + kc + BLK_K + cseg * 16,
                    &smem[(2 + nb) * TILE_BYTES + row * LDS_STRIDE + cseg * 16]);
            }
        }

        const unsigned char* aT = &smem[cur * TILE_BYTES];
        const unsigned char* bT = &smem[(2 + cur) * TILE_BYTES];

        v16i a0, a1;
        {
            const unsigned char* ap0 =
                aT + (wm * 32 + lrow) * LDS_STRIDE + half * 16;
            const unsigned char* ap1 = ap0 + 16 * LDS_STRIDE;
#pragma unroll
            for (int j = 0; j < 8; ++j) {
                uint2 d0 = *(const uint2*)(ap0 + 32 * (j >> 1) + 8 * (j & 1));
                a0[2 * j] = (int)d0.x; a0[2 * j + 1] = (int)d0.y;
                uint2 d1 = *(const uint2*)(ap1 + 32 * (j >> 1) + 8 * (j & 1));
                a1[2 * j] = (int)d1.x; a1[2 * j + 1] = (int)d1.y;
            }
        }
#pragma unroll
        for (int t = 0; t < 4; ++t) {
            v16i bF;
            const unsigned char* bp =
                bT + (wn * 64 + t * 16 + lrow) * LDS_STRIDE + half * 16;
#pragma unroll
            for (int g = 0; g < 4; ++g) {
                uint4 d = *(const uint4*)(bp + 32 * g);
                bF[4 * g]     = (int)d.x;
                bF[4 * g + 1] = (int)d.y;
                bF[4 * g + 2] = (int)d.z;
                bF[4 * g + 3] = (int)d.w;
            }
            acc[t]     = __builtin_amdgcn_wmma_f32_16x16x128_fp8_fp8(
                             a0, bF, (short)0, acc[t],     false, false);
            acc[4 + t] = __builtin_amdgcn_wmma_f32_16x16x128_fp8_fp8(
                             a1, bF, (short)0, acc[4 + t], false, false);
        }
        __syncthreads();
        cur ^= 1;
    }

    const float s = scales[1] * wscale[0];
#pragma unroll
    for (int t = 0; t < 4; ++t) {
        const int col = nblk + wn * 64 + t * 16 + lrow;
        const float bv = bias[col];
#pragma unroll
        for (int sub = 0; sub < 2; ++sub) {
#pragma unroll
            for (int r = 0; r < 8; ++r) {
                const int row = mblk + wm * 32 + sub * 16 + half * 8 + r;
                out[(size_t)row * N + col] = acc[sub * 4 + t][r] * s + bv;
            }
        }
    }
}

extern "C" void kernel_launch(void* const* d_in, const int* in_sizes, int n_in,
                              void* d_out, int out_size, void* d_ws,
                              size_t ws_size, hipStream_t stream) {
    const float* x       = (const float*)d_in[0];
    const float* qweight = (const float*)d_in[1];
    const float* w_scale = (const float*)d_in[2];
    const float* bias    = (const float*)d_in[3];
    float* out           = (float*)d_out;

    const int xsz = in_sizes[0];
    const int wsz = in_sizes[1];
    const int N   = in_sizes[3];
    const int K   = wsz / N;
    const int M   = xsz / K;

    char* ws = (char*)d_ws;
    int*   amax_bits  = (int*)ws;
    float* scales     = (float*)(ws + 16);
    unsigned char* qx = (unsigned char*)(ws + 256);
    unsigned char* qw = qx + (size_t)M * K;

    (void)hipMemsetAsync(amax_bits, 0, sizeof(int), stream);
    amax_kernel<<<512, 256, 0, stream>>>(x, xsz / 4, amax_bits);
    scale_kernel<<<1, 1, 0, stream>>>(amax_bits, scales);
    quant_kernel<<<1024, 256, 0, stream>>>(x, (unsigned int*)qx, xsz / 4,
                                           scales, 1);
    quant_kernel<<<256, 256, 0, stream>>>(qweight, (unsigned int*)qw, wsz / 4,
                                          scales, 0);

    dim3 grid(N / 128, M / 128);
    fp8_gemm_wmma_kernel<<<grid, 256, 0, stream>>>(qx, qw, scales, w_scale,
                                                   bias, out, M, N, K);
}
